// CausalAttention_61942018342958
// MI455X (gfx1250) — matrix operand order repaired and checked by reading, not compiled, not run on hardware
//
#include <hip/hip_runtime.h>
#include <hip/hip_bf16.h>

// ---------------------------------------------------------------------------
// Causal attention with RoPE, B=2 S=4096 D=1024, MI455X (gfx1250, wave32).
// Compute-bound (~190 GFLOP vs ~80MB HBM traffic) -> bf16 WMMA everywhere.
// QKV (48MB bf16) fits in 192MB L2, so Q/V fragments are re-streamed from L2;
// K tiles in the stats pass are DMA'd into LDS with the Tensor Data Mover
// (tensor_load_to_lds, TENSORcnt double-buffering); LDS also handles the
// P (probabilities) C-layout -> A-layout transpose in the output pass.
// ---------------------------------------------------------------------------

typedef __bf16 bf16;
typedef __attribute__((ext_vector_type(8)))  __bf16 v8bf;
typedef __attribute__((ext_vector_type(16))) __bf16 v16bf;
typedef __attribute__((ext_vector_type(8)))  float  v8f;
typedef __attribute__((ext_vector_type(4)))  unsigned int u32x4;
typedef __attribute__((ext_vector_type(8)))  unsigned int u32x8;

#define B_SZ   2
#define S_LEN  4096
#define D_DIM  1024
#define HALFD  512
#define KEYS_PER_PASS 2048   // P tile in LDS: 16 x 2048 bf16 = 64KB

// ---- WMMA fragment helpers (layouts per CDNA5 ISA 7.12.2, wave32) ----------

// A-matrix 16x32 bf16: lane(m = lane&15) holds row m.
// lanes 0-15 : VGPR0-3 -> K = k0+0..7,   VGPR4-7 -> K = k0+16..23
// lanes 16-31: VGPR0-3 -> K = k0+8..15,  VGPR4-7 -> K = k0+24..31
__device__ inline v16bf load_a_frag(const bf16* __restrict__ base, int ld,
                                    int k0, int lane) {
  int m = lane & 15, h = lane >> 4;
  const bf16* p = base + (size_t)m * ld + k0 + h * 8;
  v8bf lo = *(const v8bf*)p;
  v8bf hi = *(const v8bf*)(p + 16);
  return __builtin_shufflevector(lo, hi, 0,1,2,3,4,5,6,7,8,9,10,11,12,13,14,15);
}

// B-matrix 32x16 bf16: lane(n = lane&15) holds column-entity n.
// lanes 0-15: K = k0+0..7, k0+16..23; lanes 16-31: K = k0+8..15, k0+24..31.
__device__ inline v16bf load_b_frag(const bf16* __restrict__ base, int ld,
                                    int k0, int lane) {
  int n = lane & 15, h = lane >> 4;
  const bf16* p = base + (size_t)n * ld + k0 + h * 8;
  v8bf lo = *(const v8bf*)p;
  v8bf hi = *(const v8bf*)(p + 16);
  return __builtin_shufflevector(lo, hi, 0,1,2,3,4,5,6,7,8,9,10,11,12,13,14,15);
}

__device__ inline v8f wmma_bf16(v16bf a, v16bf b, v8f c) {
  // D = A x B + C, f32 accumulate. 8 args: neg_a,A,neg_b,B,c_mod,C,reuse_a,reuse_b
  return __builtin_amdgcn_wmma_f32_16x16x32_bf16(false, a, false, b,
                                                 (short)0, c, false, false);
}

// ---- Tensor Data Mover: DMA a 16 x 1024 bf16 tile (32KB) global -> LDS -----
// D# per CDNA5 ISA ch.8 (2D tensor -> 2-operand form, groups 2/3 NULL):
//  group0: dw0 = count=1; dw1 = lds_addr; dw2/dw3 = global_addr, type=2
//  group1: data_size=2B (bits17:16), tensor_dim0=1024 (bits79:48),
//          tensor_dim1=16 (bits111:80), tile_dim0=1024 (bits127:112),
//          tile_dim1=16 (bits143:128), tensor_dim0_stride=1024 (bits207:160)
// TDM ignores EXEC and issues once per wave; tracked by TENSORcnt.
__device__ inline void tdm_load_k16(const bf16* gp, unsigned lds_off) {
  unsigned long long ga = (unsigned long long)(size_t)gp;
  u32x4 g0 = { 1u, lds_off, (unsigned)ga, (unsigned)(ga >> 32) | 0x80000000u };
  u32x8 g1 = { 0x00010000u,   // workgroup_mask=0, data_size=1 (2 bytes)
               0x04000000u,   // tensor_dim0 = 1024 (low 16 in dw1[31:16])
               0x00100000u,   // tensor_dim0 hi=0, tensor_dim1 = 16
               0x04000000u,   // tensor_dim1 hi=0, tile_dim0 = 1024
               16u,           // tile_dim1 = 16, tile_dim2 = 0
               1024u,         // tensor_dim0_stride low 32
               0u, 0u };
  asm volatile("tensor_load_to_lds %0, %1" :: "s"(g0), "s"(g1) : "memory");
}

// ---- fp32 -> bf16 convert --------------------------------------------------
__global__ void cvt_f32_to_bf16(const float* __restrict__ in,
                                bf16* __restrict__ out, int n) {
  int i = (blockIdx.x * blockDim.x + threadIdx.x) * 4;
  if (i + 3 < n) {
    float4 f = *(const float4*)(in + i);
    union { bf16 b[4]; uint2 u; } r;
    r.b[0] = (bf16)f.x; r.b[1] = (bf16)f.y; r.b[2] = (bf16)f.z; r.b[3] = (bf16)f.w;
    *(uint2*)(out + i) = r.u;
  } else {
    for (; i < n; ++i) out[i] = (bf16)in[i];
  }
}

// ---- Kernel 1: fused QKV projection + RoPE ---------------------------------
// One wave per 16x16 output tile. q,k stored row-major [B*S][D] (post-RoPE);
// v stored transposed vT[b][d][s] so P*V B-fragments are contiguous later.
__global__ void qkv_rope_kernel(const bf16* __restrict__ xb,
                                const bf16* __restrict__ wq,
                                const bf16* __restrict__ wk,
                                const bf16* __restrict__ wv,
                                const float* __restrict__ fcos,
                                const float* __restrict__ fsin,
                                bf16* __restrict__ qb,
                                bf16* __restrict__ kb,
                                bf16* __restrict__ vT) {
  int lane = threadIdx.x & 31;
  int wave = threadIdx.x >> 5;
  int tile = blockIdx.x * 4 + wave;          // 512 row-tiles x 64 col-tiles
  int rt = tile >> 6;
  int ct = tile & 63;
  int rowbase = rt * 16;
  int colbase = ct * 16;

  v8f aq = {}, ak = {}, av = {};
  const bf16* Abase = xb + (size_t)rowbase * D_DIM;
  const bf16* Bq = wq + (size_t)colbase * D_DIM;
  const bf16* Bk = wk + (size_t)colbase * D_DIM;
  const bf16* Bv = wv + (size_t)colbase * D_DIM;
  for (int k0 = 0; k0 < D_DIM; k0 += 32) {
    v16bf a  = load_a_frag(Abase, D_DIM, k0, lane);
    aq = wmma_bf16(a, load_b_frag(Bq, D_DIM, k0, lane), aq);
    ak = wmma_bf16(a, load_b_frag(Bk, D_DIM, k0, lane), ak);
    av = wmma_bf16(a, load_b_frag(Bv, D_DIM, k0, lane), av);
  }

  // RoPE on q,k accumulators. C layout: VGPR r holds row (r + 8*half),
  // column = colbase + (lane&15). Pair partner column differs by 1 -> shfl_xor.
  int h = lane >> 4;
  int n = colbase + (lane & 15);
  int pair = n >> 1;
  bool odd = (n & 1) != 0;
#pragma unroll
  for (int r = 0; r < 8; ++r) {
    int row = rowbase + r + h * 8;           // flat over B*S
    int s   = row & (S_LEN - 1);
    float cs = fcos[(size_t)s * HALFD + pair];
    float sn = fsin[(size_t)s * HALFD + pair];
    float q = aq[r], k = ak[r];
    float qp = __shfl_xor(q, 1, 32);
    float kp = __shfl_xor(k, 1, 32);
    float qo = odd ? (qp * sn + q * cs) : (q * cs - qp * sn);
    float ko = odd ? (kp * sn + k * cs) : (k * cs - kp * sn);
    qb[(size_t)row * D_DIM + n] = (bf16)qo;
    kb[(size_t)row * D_DIM + n] = (bf16)ko;
    int b = row >> 12;                       // 4096 rows per batch
    vT[((size_t)b * D_DIM + n) * S_LEN + s] = (bf16)av[r];
  }
}

// ---- Kernel 2: flash stats pass (row max m, row sumexp l) ------------------
// One wave per query tile. K tiles are DMA'd into LDS by the TDM with
// double buffering: issue tile j+1, wait TENSORcnt<=1 (tile j landed),
// compute scores for tile j from LDS while tile j+1 streams in.
__global__ void attn_stats_kernel(const bf16* __restrict__ qb,
                                  const bf16* __restrict__ kb,
                                  float* __restrict__ mbuf,
                                  float* __restrict__ lbuf) {
  __shared__ __attribute__((aligned(128))) bf16 Ksh[2][16 * D_DIM]; // 64KB
  int lane = threadIdx.x & 31;
  int qt = blockIdx.x;                       // 0..511
  int b = qt >> 8;
  int it = qt & 255;
  const bf16* Q  = qb + ((size_t)b * S_LEN + (size_t)it * 16) * D_DIM;
  const bf16* K0 = kb + (size_t)b * S_LEN * D_DIM;
  int h = lane >> 4;
  int ncol = lane & 15;
  unsigned lds0 = (unsigned)(size_t)&Ksh[0][0];   // generic low 32 = LDS offset
  unsigned lds1 = (unsigned)(size_t)&Ksh[1][0];

  float m[8], l[8];
#pragma unroll
  for (int r = 0; r < 8; ++r) { m[r] = -3.0e38f; l[r] = 0.0f; }
  const float sc = 0.03125f * 1.44269504089f;   // 1/sqrt(1024) * log2(e)

  tdm_load_k16(K0, lds0);                       // prime buffer 0 with tile 0
  for (int j = 0; j <= it; ++j) {
    if (j < it) {
      tdm_load_k16(K0 + (size_t)(j + 1) * 16 * D_DIM, (j & 1) ? lds0 : lds1);
      asm volatile("s_wait_tensorcnt 0x1" ::: "memory");  // tile j is resident
    } else {
      asm volatile("s_wait_tensorcnt 0x0" ::: "memory");
    }
    const bf16* Kt = &Ksh[j & 1][0];
    v8f sacc = {};
    for (int k0 = 0; k0 < D_DIM; k0 += 32) {
      sacc = wmma_bf16(load_a_frag(Q, D_DIM, k0, lane),
                       load_b_frag(Kt, D_DIM, k0, lane), sacc);
    }
#pragma unroll
    for (int r = 0; r < 8; ++r) {
      int row = r + h * 8;
      float t = sacc[r] * sc;
      bool masked = (j == it) && (ncol > row);
      if (masked) t = -3.0e38f;
      float tmax = t;
#pragma unroll
      for (int off = 1; off < 16; off <<= 1)
        tmax = fmaxf(tmax, __shfl_xor(tmax, off, 32));
      float mnew = fmaxf(m[r], tmax);
      float p = masked ? 0.0f : __builtin_amdgcn_exp2f(t - mnew);
      float psum = p;
#pragma unroll
      for (int off = 1; off < 16; off <<= 1)
        psum += __shfl_xor(psum, off, 32);
      l[r] = l[r] * __builtin_amdgcn_exp2f(m[r] - mnew) + psum;
      m[r] = mnew;
    }
  }
  if (ncol == 0) {
#pragma unroll
    for (int r = 0; r < 8; ++r) {
      int row = b * S_LEN + it * 16 + r + h * 8;
      mbuf[row] = m[r];
      lbuf[row] = l[r];
    }
  }
}

// ---- Kernel 3: output pass -------------------------------------------------
// One wave / 16-query tile. Per 2048-key pass: recompute scores (WMMA),
// normalize with precomputed stats, stage P bf16 in LDS (C->A transpose),
// then O_c += P x V via WMMA for all 64 dim chunks. V read pre-transposed.
__global__ void attn_out_kernel(const bf16* __restrict__ qb,
                                const bf16* __restrict__ kb,
                                const bf16* __restrict__ vT,
                                const float* __restrict__ mbuf,
                                const float* __restrict__ lbuf,
                                float* __restrict__ out) {
  __shared__ __attribute__((aligned(64))) bf16 Psh[16 * KEYS_PER_PASS]; // 64KB
  int lane = threadIdx.x & 31;
  int qt = blockIdx.x;
  int b = qt >> 8;
  int it = qt & 255;
  const bf16* Q  = qb + ((size_t)b * S_LEN + (size_t)it * 16) * D_DIM;
  const bf16* K0 = kb + (size_t)b * S_LEN * D_DIM;
  const bf16* V0 = vT + (size_t)b * D_DIM * S_LEN;
  int h = lane >> 4;
  int ncol = lane & 15;
  const float sc = 0.03125f * 1.44269504089f;

  float mrow[8], linv[8];
#pragma unroll
  for (int r = 0; r < 8; ++r) {
    int row = b * S_LEN + it * 16 + r + h * 8;
    mrow[r] = mbuf[row];
    linv[r] = 1.0f / lbuf[row];
  }

  int ntiles = it + 1;                        // valid 16-key tiles (causal)
  for (int pass = 0; pass * 128 < ntiles; ++pass) {
    int nv = ntiles - pass * 128; if (nv > 128) nv = 128;
    int npairs = (nv + 1) >> 1;               // 32-key pairs staged in LDS

    // Phase A: scores -> normalized P in LDS (row-major [16][2048] bf16)
    for (int jl = 0; jl < npairs * 2; ++jl) {
      int j = pass * 128 + jl;
      if (j < ntiles) {
        const bf16* Kt = K0 + (size_t)j * 16 * D_DIM;
        if (j < it) __builtin_prefetch(Kt + 16 * D_DIM, 0, 0);
        v8f sacc = {};
        for (int k0 = 0; k0 < D_DIM; k0 += 32) {
          sacc = wmma_bf16(load_a_frag(Q, D_DIM, k0, lane),
                           load_b_frag(Kt, D_DIM, k0, lane), sacc);
        }
#pragma unroll
        for (int r = 0; r < 8; ++r) {
          int row = r + h * 8;
          float p = __builtin_amdgcn_exp2f(sacc[r] * sc - mrow[r]) * linv[r];
          if (j == it && ncol > row) p = 0.0f;
          Psh[row * KEYS_PER_PASS + jl * 16 + ncol] = (bf16)p;
        }
      } else {                                 // zero-pad odd trailing tile
#pragma unroll
        for (int r = 0; r < 8; ++r)
          Psh[(r + h * 8) * KEYS_PER_PASS + jl * 16 + ncol] = (bf16)0.0f;
      }
    }
    __syncthreads();

    // Phase B: O_c += P x V for every 16-wide dim chunk
    for (int c = 0; c < 64; ++c) {
      v8f oacc;
      if (pass == 0) {
        oacc = (v8f){};
      } else {
#pragma unroll
        for (int r = 0; r < 8; ++r)
          oacc[r] = out[((size_t)(b * S_LEN + it * 16 + r + h * 8)) * D_DIM
                        + c * 16 + ncol];
      }
      const bf16* Vp = V0 + (size_t)c * 16 * S_LEN + (size_t)pass * KEYS_PER_PASS;
      for (int jp = 0; jp < npairs; ++jp) {
        v16bf a  = load_a_frag(Psh, KEYS_PER_PASS, jp * 32, lane);
        v16bf bb = load_b_frag(Vp, S_LEN, jp * 32, lane);
        oacc = wmma_bf16(a, bb, oacc);
      }
#pragma unroll
      for (int r = 0; r < 8; ++r)
        out[((size_t)(b * S_LEN + it * 16 + r + h * 8)) * D_DIM
            + c * 16 + ncol] = oacc[r];
    }
    __syncthreads();
  }
}

// ---------------------------------------------------------------------------
extern "C" void kernel_launch(void* const* d_in, const int* in_sizes, int n_in,
                              void* d_out, int out_size, void* d_ws, size_t ws_size,
                              hipStream_t stream) {
  const float* x    = (const float*)d_in[0];
  const float* w_q  = (const float*)d_in[1];
  const float* w_k  = (const float*)d_in[2];
  const float* w_v  = (const float*)d_in[3];
  const float* fcos = (const float*)d_in[4];
  const float* fsin = (const float*)d_in[5];
  float* out = (float*)d_out;

  // Workspace partition (~70 MB total)
  char* ws = (char*)d_ws;
  size_t off = 0;
  auto alloc = [&](size_t bytes) -> void* {
    void* p = ws + off;
    off = (off + bytes + 255) & ~(size_t)255;
    return p;
  };
  const size_t NX = (size_t)B_SZ * S_LEN * D_DIM;   // 8,388,608
  const size_t NW = (size_t)D_DIM * D_DIM;          // 1,048,576
  bf16* xb  = (bf16*)alloc(NX * 2);
  bf16* wqb = (bf16*)alloc(NW * 2);
  bf16* wkb = (bf16*)alloc(NW * 2);
  bf16* wvb = (bf16*)alloc(NW * 2);
  bf16* qb  = (bf16*)alloc(NX * 2);
  bf16* kb  = (bf16*)alloc(NX * 2);
  bf16* vT  = (bf16*)alloc(NX * 2);
  float* mbuf = (float*)alloc((size_t)B_SZ * S_LEN * 4);
  float* lbuf = (float*)alloc((size_t)B_SZ * S_LEN * 4);

  // 1) fp32 -> bf16 converts
  cvt_f32_to_bf16<<<(int)(NX / 1024), 256, 0, stream>>>(x,  xb,  (int)NX);
  cvt_f32_to_bf16<<<(int)(NW / 1024), 256, 0, stream>>>(w_q, wqb, (int)NW);
  cvt_f32_to_bf16<<<(int)(NW / 1024), 256, 0, stream>>>(w_k, wkb, (int)NW);
  cvt_f32_to_bf16<<<(int)(NW / 1024), 256, 0, stream>>>(w_v, wvb, (int)NW);

  // 2) fused QKV projection + RoPE: 512*64 tiles, 4 waves/block
  qkv_rope_kernel<<<8192, 128, 0, stream>>>(xb, wqb, wkb, wvb, fcos, fsin,
                                            qb, kb, vT);

  // 3) softmax stats: 1 wave per query tile, TDM-staged K tiles (64KB LDS)
  attn_stats_kernel<<<512, 32, 0, stream>>>(qb, kb, mbuf, lbuf);

  // 4) output pass: 1 wave per query tile, 64KB LDS each
  attn_out_kernel<<<512, 32, 0, stream>>>(qb, kb, vT, mbuf, lbuf, out);

  (void)in_sizes; (void)n_in; (void)out_size; (void)ws_size;
}
